// VAEAttention_34394098106447
// MI455X (gfx1250) — matrix operand order repaired and checked by reading, not compiled, not run on hardware
//
#include <hip/hip_runtime.h>

typedef __attribute__((ext_vector_type(16))) _Float16 v16h;
typedef __attribute__((ext_vector_type(8)))  _Float16 v8h;
typedef __attribute__((ext_vector_type(8)))  float    v8f;
typedef __attribute__((ext_vector_type(4)))  float    v4f;
typedef __attribute__((ext_vector_type(4)))  int      i4;

#define C_DIM 512
#define N_TOK 4096
#define B_DIM 4
#define GROUPS 32
#define CPG 16            // C / GROUPS
#define STRIPE 1024       // query rows per attention stripe

// Async global->LDS (ASYNCcnt) path if the toolchain exposes the builtins.
// Probe-confirmed signature: (int4 AS(1)*, int4 AS(3)*, imm offset, imm cpol).
#if defined(__AMDGCN__) && \
    __has_builtin(__builtin_amdgcn_global_load_async_to_lds_b128) && \
    __has_builtin(__builtin_amdgcn_s_wait_asynccnt)
#define USE_ASYNC_LDS 1
#define ASYNC_B128(gp, lp)                                             \
  __builtin_amdgcn_global_load_async_to_lds_b128(                      \
      (__attribute__((address_space(1))) i4*)(const void*)(gp),        \
      (__attribute__((address_space(3))) i4*)(void*)(lp), 0, 0)
#else
#define USE_ASYNC_LDS 0
#endif

// ---------------------------------------------------------------------------
// GroupNorm statistics: one block per (batch, group); reduce 4096*16 floats.
// ---------------------------------------------------------------------------
__global__ __launch_bounds__(256) void gn_stats(const float* __restrict__ x,
                                                float* __restrict__ stats) {
  const int b = blockIdx.x >> 5;
  const int g = blockIdx.x & 31;
  const float* base = x + (size_t)b * N_TOK * C_DIM + g * CPG;
  const int tid = threadIdx.x;
  float s = 0.f, ss = 0.f;
  for (int n = tid; n < N_TOK; n += 256) {
    const float* pp = base + (size_t)n * C_DIM;
#pragma unroll
    for (int k = 0; k < CPG; k += 4) {
      v4f v = *(const v4f*)(pp + k);
#pragma unroll
      for (int e = 0; e < 4; ++e) { s += v[e]; ss += v[e] * v[e]; }
    }
  }
  __shared__ float r1[256], r2[256];
  r1[tid] = s; r2[tid] = ss;
  __syncthreads();
  for (int st = 128; st > 0; st >>= 1) {
    if (tid < st) { r1[tid] += r1[tid + st]; r2[tid] += r2[tid + st]; }
    __syncthreads();
  }
  if (tid == 0) {
    const float inv_cnt = 1.f / (float)(N_TOK * CPG);
    float mean = r1[0] * inv_cnt;
    float var  = r2[0] * inv_cnt - mean * mean;
    stats[2 * blockIdx.x]     = mean;
    stats[2 * blockIdx.x + 1] = rsqrtf(var + 1e-6f);
  }
}

// ---------------------------------------------------------------------------
// Apply GroupNorm affine, emit f16 activations. 4 elements / thread.
// ---------------------------------------------------------------------------
__global__ __launch_bounds__(256) void gn_apply(const float* __restrict__ x,
                                                const float* __restrict__ stats,
                                                const float* __restrict__ gamma,
                                                const float* __restrict__ beta,
                                                _Float16* __restrict__ xn) {
  size_t base = ((size_t)blockIdx.x * 256 + threadIdx.x) * 4;
  const int c = (int)(base & (C_DIM - 1));
  const int b = (int)(base >> 21);           // / (512*4096)
  const int g = c >> 4;
  const float mean = stats[2 * (b * GROUPS + g)];
  const float rstd = stats[2 * (b * GROUPS + g) + 1];
  v4f v = *(const v4f*)(x + base);
#pragma unroll
  for (int k = 0; k < 4; ++k) {
    float xv = (v[k] - mean) * rstd * gamma[c + k] + beta[c + k];
    xn[base + k] = (_Float16)xv;
  }
}

// ---------------------------------------------------------------------------
// Transpose the 4 weight matrices (f32 [C][D]) into f16 N-major [D][C].
// ---------------------------------------------------------------------------
__global__ __launch_bounds__(256) void wtrans(const float* __restrict__ W0,
                                              const float* __restrict__ W1,
                                              const float* __restrict__ W2,
                                              const float* __restrict__ W3,
                                              _Float16* __restrict__ T) {
  const float* Ws[4] = {W0, W1, W2, W3};
  const float* W = Ws[blockIdx.z];
  _Float16* Wt = T + (size_t)blockIdx.z * C_DIM * C_DIM;
  __shared__ float t[32][33];
  const int tx = threadIdx.x, ty = threadIdx.y;   // 32 x 8
  const int c0 = blockIdx.y * 32, d0 = blockIdx.x * 32;
#pragma unroll
  for (int i = 0; i < 4; ++i)
    t[ty + 8 * i][tx] = W[(size_t)(c0 + ty + 8 * i) * C_DIM + d0 + tx];
  __syncthreads();
#pragma unroll
  for (int i = 0; i < 4; ++i)
    Wt[(size_t)(d0 + ty + 8 * i) * C_DIM + c0 + tx] = (_Float16)t[tx][ty + 8 * i];
}

// ---------------------------------------------------------------------------
// Row softmax: one block per row of N floats -> f16 probabilities.
// ---------------------------------------------------------------------------
__global__ __launch_bounds__(256) void softmax_rows(const float* __restrict__ S,
                                                    _Float16* __restrict__ P,
                                                    int N) {
  __shared__ float red[256];
  const int row = blockIdx.x;
  const float* s = S + (size_t)row * N;
  _Float16* p = P + (size_t)row * N;
  const int tid = threadIdx.x;
  float mx = -3.4e38f;
  for (int i = tid; i < N; i += 256) mx = fmaxf(mx, s[i]);
  red[tid] = mx; __syncthreads();
  for (int st = 128; st > 0; st >>= 1) {
    if (tid < st) red[tid] = fmaxf(red[tid], red[tid + st]);
    __syncthreads();
  }
  mx = red[0]; __syncthreads();
  float sum = 0.f;
  for (int i = tid; i < N; i += 256) {
    float e = __expf(s[i] - mx);
    sum += e;
    p[i] = (_Float16)e;
  }
  red[tid] = sum; __syncthreads();
  for (int st = 128; st > 0; st >>= 1) {
    if (tid < st) red[tid] += red[tid + st];
    __syncthreads();
  }
  const float inv = 1.f / red[0];
  for (int i = tid; i < N; i += 256) p[i] = (_Float16)((float)p[i] * inv);
}

// ---------------------------------------------------------------------------
// WMMA GEMM: Out[m,n] = sum_k A[m,k] * Bsrc[n,k]   (Bsrc is N-major)
// Block tile 128x128, BK=64, LDS double-buffered. 8 waves as 4(M) x 2(N);
// each wave owns a 32x64 subtile = 2x4 accumulators; 16 WMMA per K-chunk.
// Global->LDS uses async-to-LDS (ASYNCcnt) when available.
// ---------------------------------------------------------------------------
enum GemmMode { OUT_F16 = 0, OUT_F16_TRANS = 1, OUT_F32 = 2, OUT_F32_RESID = 3 };

__device__ inline v16h cat8(v8h lo, v8h hi) {
  return __builtin_shufflevector(lo, hi, 0, 1, 2, 3, 4, 5, 6, 7,
                                 8, 9, 10, 11, 12, 13, 14, 15);
}

template <int MODE>
__global__ __launch_bounds__(256) void gemm_f16(
    const _Float16* __restrict__ A, int lda,
    const _Float16* __restrict__ Bsrc, int ldb,
    void* __restrict__ Out, int ldo,
    const float* __restrict__ bias,
    const float* __restrict__ resid,
    float scale, int Ktot) {
  constexpr int BM = 128, BN = 128, BK = 64, PAD = 8;
  constexpr int LDR = BK + PAD;                 // 72 halfs = 144 B (16B multiple)
  __shared__ _Float16 Al[2][BM][LDR];
  __shared__ _Float16 Bl[2][BN][LDR];

  const int tid  = threadIdx.x;
  const int lane = tid & 31;
  const int wave = tid >> 5;
  const int wm = wave & 3;        // 4 waves along M
  const int wn = wave >> 2;       // 2 waves along N
  const int m0 = blockIdx.y * BM;
  const int n0 = blockIdx.x * BN;

  v8f acc[2][4] = {};

  // global->LDS: each thread owns one 32-half (64B) row segment of A and B
  const int ld_row = tid >> 1;          // 0..127
  const int ld_off = (tid & 1) * 32;    // 0 or 32
  const _Float16* Ag = A    + (size_t)(m0 + ld_row) * lda + ld_off;
  const _Float16* Bg = Bsrc + (size_t)(n0 + ld_row) * ldb + ld_off;

  // fragment addressing (CDNA5 wave32 WMMA layouts)
  const int hk  = (lane >> 4) * 8;      // A: K sub-offset per half-wave
  const int rA  = lane & 15;            // A: row within 16x16 tile
  const int bko = (lane >> 4) * 8;      // B: K sub-offset per half-wave
  const int nl  = lane & 15;            // B/D: column within tile

#if USE_ASYNC_LDS
  // preload chunk 0 into buffer 0 (direct DMA into LDS, no VGPR staging)
#pragma unroll
  for (int u = 0; u < 4; ++u) {
    ASYNC_B128(Ag + u * 8, &Al[0][ld_row][ld_off + u * 8]);
    ASYNC_B128(Bg + u * 8, &Bl[0][ld_row][ld_off + u * 8]);
  }
  __builtin_amdgcn_s_wait_asynccnt(0);
  __syncthreads();

  int p = 0;
  for (int kc = 0; kc < Ktot; kc += BK) {
    const bool more = (kc + BK) < Ktot;
    if (more) {  // DMA next chunk into the inactive buffer
#pragma unroll
      for (int u = 0; u < 4; ++u) {
        ASYNC_B128(Ag + kc + BK + u * 8, &Al[p ^ 1][ld_row][ld_off + u * 8]);
        ASYNC_B128(Bg + kc + BK + u * 8, &Bl[p ^ 1][ld_row][ld_off + u * 8]);
      }
    }
#else
  v8h ar[4], br[4];
  // preload chunk 0 into buffer 0
#pragma unroll
  for (int u = 0; u < 4; ++u) {
    ar[u] = *(const v8h*)(Ag + u * 8);
    br[u] = *(const v8h*)(Bg + u * 8);
  }
#pragma unroll
  for (int u = 0; u < 4; ++u) {
    *(v8h*)&Al[0][ld_row][ld_off + u * 8] = ar[u];
    *(v8h*)&Bl[0][ld_row][ld_off + u * 8] = br[u];
  }
  __syncthreads();

  int p = 0;
  for (int kc = 0; kc < Ktot; kc += BK) {
    const bool more = (kc + BK) < Ktot;
    if (more) {  // issue next-chunk global loads before the math
#pragma unroll
      for (int u = 0; u < 4; ++u) {
        ar[u] = *(const v8h*)(Ag + kc + BK + u * 8);
        br[u] = *(const v8h*)(Bg + kc + BK + u * 8);
      }
    }
#endif
#pragma unroll
    for (int ks = 0; ks < BK; ks += 32) {
      v16h af[2], bf[4];
#pragma unroll
      for (int i = 0; i < 2; ++i) {
        const _Float16* ap = &Al[p][wm * 32 + i * 16 + rA][ks];
        af[i] = cat8(*(const v8h*)(ap + hk), *(const v8h*)(ap + 16 + hk));
      }
#pragma unroll
      for (int j = 0; j < 4; ++j) {
        const _Float16* bp = &Bl[p][wn * 64 + j * 16 + nl][ks + bko];
        bf[j] = cat8(*(const v8h*)bp, *(const v8h*)(bp + 16));
      }
#pragma unroll
      for (int i = 0; i < 2; ++i)
#pragma unroll
        for (int j = 0; j < 4; ++j)
          acc[i][j] = __builtin_amdgcn_wmma_f32_16x16x32_f16(
              false, af[i], false, bf[j], (short)0, acc[i][j], false, false);
    }
#if USE_ASYNC_LDS
    if (more) __builtin_amdgcn_s_wait_asynccnt(0);
#else
    if (more) {  // stores target the inactive buffer: no race with readers
#pragma unroll
      for (int u = 0; u < 4; ++u) {
        *(v8h*)&Al[p ^ 1][ld_row][ld_off + u * 8] = ar[u];
        *(v8h*)&Bl[p ^ 1][ld_row][ld_off + u * 8] = br[u];
      }
    }
#endif
    __syncthreads();
    p ^= 1;
  }

  // epilogue: C/D layout => lane holds N=lane%16, M=r + 8*(lane/16)
  const int moff = (lane >> 4) * 8;
#pragma unroll
  for (int i = 0; i < 2; ++i)
#pragma unroll
    for (int j = 0; j < 4; ++j) {
      const int ncol = n0 + wn * 64 + j * 16 + nl;
      const float bv = bias ? bias[ncol] : 0.f;
#pragma unroll
      for (int r = 0; r < 8; ++r) {
        const int mrow = m0 + wm * 32 + i * 16 + moff + r;
        const float val = (acc[i][j][r] + bv) * scale;
        if (MODE == OUT_F16) {
          ((_Float16*)Out)[(size_t)mrow * ldo + ncol] = (_Float16)val;
        } else if (MODE == OUT_F16_TRANS) {
          // V^T: per-batch [C][N_TOK] blocks
          ((_Float16*)Out)[(size_t)(mrow >> 12) * (C_DIM * N_TOK) +
                           (size_t)ncol * ldo + (mrow & (N_TOK - 1))] =
              (_Float16)val;
        } else if (MODE == OUT_F32) {
          ((float*)Out)[(size_t)mrow * ldo + ncol] = val;
        } else {
          ((float*)Out)[(size_t)mrow * ldo + ncol] =
              val + resid[(size_t)mrow * ldo + ncol];
        }
      }
    }
}

// ---------------------------------------------------------------------------
extern "C" void kernel_launch(void* const* d_in, const int* in_sizes, int n_in,
                              void* d_out, int out_size, void* d_ws, size_t ws_size,
                              hipStream_t stream) {
  const float* x     = (const float*)d_in[0];
  const float* gamma = (const float*)d_in[1];
  const float* beta  = (const float*)d_in[2];
  const float* wq = (const float*)d_in[3]; const float* bq = (const float*)d_in[4];
  const float* wk = (const float*)d_in[5]; const float* bk = (const float*)d_in[6];
  const float* wv = (const float*)d_in[7]; const float* bv = (const float*)d_in[8];
  const float* wo = (const float*)d_in[9]; const float* bo = (const float*)d_in[10];
  float* out = (float*)d_out;

  char* wsb = (char*)d_ws;
  size_t off = 0;
  auto alloc = [&](size_t bytes) -> void* {
    void* p = wsb + off;
    off = (off + bytes + 255) & ~(size_t)255;
    return p;
  };
  const size_t ROWS = (size_t)B_DIM * N_TOK;                // 16384
  float*    stats = (float*)   alloc(B_DIM * GROUPS * 2 * sizeof(float));
  _Float16* wT    = (_Float16*)alloc(4 * (size_t)C_DIM * C_DIM * 2);
  _Float16* xn    = (_Float16*)alloc(ROWS * C_DIM * 2);
  _Float16* Q     = (_Float16*)alloc(ROWS * C_DIM * 2);
  _Float16* Kf    = (_Float16*)alloc(ROWS * C_DIM * 2);
  _Float16* Vt    = (_Float16*)alloc(ROWS * C_DIM * 2);     // per-batch [C][N]
  float*    Sc    = (float*)   alloc((size_t)STRIPE * N_TOK * 4);
  _Float16* Pa    = (_Float16*)alloc((size_t)STRIPE * N_TOK * 2);
  _Float16* AO    = (_Float16*)alloc(ROWS * C_DIM * 2);

  gn_stats<<<B_DIM * GROUPS, 256, 0, stream>>>(x, stats);
  gn_apply<<<(int)(ROWS * C_DIM / (4 * 256)), 256, 0, stream>>>(x, stats, gamma, beta, xn);
  wtrans<<<dim3(16, 16, 4), dim3(32, 8), 0, stream>>>(wq, wk, wv, wo, wT);

  const float qscale = 0.04419417382415922f;  // 1/sqrt(512)
  const size_t WSTEP = (size_t)C_DIM * C_DIM;

  // projections: Out[m,d] = sum_c xn[m,c] * w[c,d] (+b)
  gemm_f16<OUT_F16>      <<<dim3(4, 128), 256, 0, stream>>>(xn, C_DIM, wT + 0 * WSTEP, C_DIM, Q,  C_DIM, bq, nullptr, qscale, C_DIM);
  gemm_f16<OUT_F16>      <<<dim3(4, 128), 256, 0, stream>>>(xn, C_DIM, wT + 1 * WSTEP, C_DIM, Kf, C_DIM, bk, nullptr, 1.f,    C_DIM);
  gemm_f16<OUT_F16_TRANS><<<dim3(4, 128), 256, 0, stream>>>(xn, C_DIM, wT + 2 * WSTEP, C_DIM, Vt, N_TOK, bv, nullptr, 1.f,    C_DIM);

  for (int b = 0; b < B_DIM; ++b)
    for (int s = 0; s < N_TOK / STRIPE; ++s) {
      const _Float16* Aq = Q  + ((size_t)b * N_TOK + s * STRIPE) * C_DIM;
      const _Float16* Bk = Kf + (size_t)b * N_TOK * C_DIM;
      // scores stripe: [STRIPE x N_TOK], K-dim = C
      gemm_f16<OUT_F32><<<dim3(N_TOK / 128, STRIPE / 128), 256, 0, stream>>>(
          Aq, C_DIM, Bk, C_DIM, Sc, N_TOK, nullptr, nullptr, 1.f, C_DIM);
      softmax_rows<<<STRIPE, 256, 0, stream>>>(Sc, Pa, N_TOK);
      const _Float16* Bv = Vt + (size_t)b * C_DIM * N_TOK;
      _Float16* Oo = AO + ((size_t)b * N_TOK + s * STRIPE) * C_DIM;
      // attn @ V: [STRIPE x C], K-dim = N_TOK, B-source is V^T (N-major)
      gemm_f16<OUT_F16><<<dim3(C_DIM / 128, STRIPE / 128), 256, 0, stream>>>(
          Pa, N_TOK, Bv, N_TOK, Oo, C_DIM, nullptr, nullptr, 1.f, N_TOK);
    }

  // final projection + bias + residual
  gemm_f16<OUT_F32_RESID><<<dim3(4, 128), 256, 0, stream>>>(
      AO, C_DIM, wT + 3 * WSTEP, C_DIM, out, C_DIM, bo, x, 1.f, C_DIM);
}
